// Module_GAT_CS_split_49254684950644
// MI455X (gfx1250) — matrix operand order repaired and checked by reading, not compiled, not run on hardware
//
#include <hip/hip_runtime.h>
#include <hip/hip_bf16.h>

// ---------------------------------------------------------------------------
// MI455X (gfx1250) implementation of the GAT_CS_split reference.
//
// Sizing: B=2, N=1024, D_IN=10, D_ATT=32, att hidden=64, 6 coords.
// Dominant cost: per-pair mish(K'[i]+Q'[j]) (134M mish) + two pairwise GEMMs
// ([pairs x 64]@[64x6] scores, [16 x N]@[N x 32] agg) -> v_wmma_f32_16x16x32_f16.
// Everything else is algebraically collapsed (see analysis).
// ---------------------------------------------------------------------------

typedef __attribute__((ext_vector_type(16))) _Float16 v16h;
typedef __attribute__((ext_vector_type(8)))  float    v8f;

#define Bsz   2
#define Nsz   1024
#define BN    (Bsz * Nsz)
#define DH    64      // att hidden
#define DA    32      // D_ATT
#define NC    6       // coords

__device__ __forceinline__ float mishf(float x) {
  // x * tanh(softplus(x)) == x * ((1+e^x)^2 - 1) / ((1+e^x)^2 + 1)
  float e = __expf(fminf(x, 20.0f));
  float u = 1.0f + e;
  float n = u * u;
  return x * (n - 1.0f) / (n + 1.0f);
}

__device__ __forceinline__ float sigmoidf_(float x) {
  return 1.0f / (1.0f + __expf(-x));
}

// -------------------------- parameter bundles ------------------------------
struct EncParams {
  const float *x;
  const float *kW1, *kB1, *kW2, *kB2;
  const float *qW1, *qB1, *qW2, *qB2;
  const float *vW1, *vB1, *vW2, *vB2;
  const float *aW1, *aB1;      // att layer 1 [64,64], bias [64]
};

struct DecParams {
  const float *decW[NC], *decB[NC];
  const float *muW[NC],  *muB[NC];
  const float *sigW[NC], *sigB[NC];
};

// ---------------------------------------------------------------------------
// Kernel 1: token MLPs k,q,v ([10->64 mish ->32]) + factorized att layer 1:
//   K'[t] = k@aW1[0:32,:] + aB1   (f16)
//   Q'[t] = q@aW1[32:64,:]        (f16)
// Also emits v as f32 (decoder) and v transposed as f16 (WMMA B operand).
// ---------------------------------------------------------------------------
__device__ void enc_mlp(const float* __restrict__ x10,
                        const float* __restrict__ W1, const float* __restrict__ b1,
                        const float* __restrict__ W2, const float* __restrict__ b2,
                        float* out32) {
  float h[64];
#pragma unroll 4
  for (int o = 0; o < 64; ++o) {
    float s = b1[o];
#pragma unroll
    for (int i = 0; i < 10; ++i) s += x10[i] * W1[i * 64 + o];
    h[o] = mishf(s);
  }
#pragma unroll 4
  for (int o = 0; o < 32; ++o) {
    float s = b2[o];
#pragma unroll 8
    for (int k = 0; k < 64; ++k) s += h[k] * W2[k * 32 + o];
    out32[o] = s;
  }
}

__global__ __launch_bounds__(256) void enc_kernel(
    EncParams P,
    _Float16* __restrict__ kp, _Float16* __restrict__ qp,
    _Float16* __restrict__ vh, float* __restrict__ vbuf) {
  const int t = blockIdx.x * 256 + threadIdx.x;
  if (t >= BN) return;
  const int b = t / Nsz, n = t % Nsz;

  float x10[10];
#pragma unroll
  for (int i = 0; i < 10; ++i) x10[i] = P.x[(size_t)t * 10 + i];

  float k32[32], q32[32], v32[32];
  enc_mlp(x10, P.kW1, P.kB1, P.kW2, P.kB2, k32);
  enc_mlp(x10, P.qW1, P.qB1, P.qW2, P.qB2, q32);
  enc_mlp(x10, P.vW1, P.vB1, P.vW2, P.vB2, v32);

#pragma unroll 4
  for (int d = 0; d < 32; ++d) {
    vbuf[(size_t)t * 32 + d] = v32[d];
    vh[((size_t)b * 32 + d) * Nsz + n] = (_Float16)v32[d];   // transposed f16
  }

#pragma unroll 2
  for (int o = 0; o < 64; ++o) {
    float sk = P.aB1[o], sq = 0.0f;
#pragma unroll 8
    for (int d = 0; d < 32; ++d) {
      sk += k32[d] * P.aW1[d * 64 + o];           // rows 0..31 -> k side
      sq += q32[d] * P.aW1[(32 + d) * 64 + o];    // rows 32..63 -> q side
    }
    kp[(size_t)t * 64 + o] = (_Float16)sk;
    qp[(size_t)t * 64 + o] = (_Float16)sq;
  }
}

// ---------------------------------------------------------------------------
// Kernel 2: fused pairwise stage. One block = (b, 16-row i-tile). 8 waves,
// each wave owns 4 j-chunks of 32. Per j: mish A-fragment built directly in
// the 16-bit A-matrix VGPR pattern, 2x WMMA vs W2 -> scores; sigmoid+mask ->
// per-wave LDS; per chunk: 6 coords x 2 d-tiles WMMA vs V -> f32 accumulators;
// cross-wave reduction via ds_add_f32.
// ---------------------------------------------------------------------------
__global__ __launch_bounds__(256) void attn_kernel(
    const _Float16* __restrict__ kp, const _Float16* __restrict__ qp,
    const _Float16* __restrict__ vh,
    const float* __restrict__ aW2,    // [64,6] row-major
    const float* __restrict__ aB2,    // [6]
    float* __restrict__ agg_g) {      // [6][B*N][32]
  __shared__ _Float16 wlds[8][NC][16][33];   // per-wave w staging (~50.7 KB)
  __shared__ float    aggL[NC][16][32];      // cross-wave reduction (12 KB)

  const int tid    = threadIdx.x;
  const int wave   = tid >> 5;
  const int lane   = tid & 31;
  const int laneHi = lane >> 4;     // 0/1: which half-wave
  const int laneLo = lane & 15;
  const int wg = blockIdx.x;
  const int b  = wg >> 6;           // 64 i-tiles per batch
  const int i0 = (wg & 63) << 4;

  for (int idx = tid; idx < NC * 16 * 32; idx += 256)
    ((float*)aggL)[idx] = 0.0f;
  __syncthreads();

  const int r0 = laneHi ? 8 : 0;    // A-fragment K-run starts
  const int r1 = laneHi ? 24 : 16;

  // Loop-invariant K' A-fragments (M = i = laneLo, K = hidden d).
  v16h aK0, aK1;
  {
    const _Float16* p = kp + (size_t)(b * Nsz + i0 + laneLo) * 64;
#pragma unroll
    for (int h = 0; h < 8; ++h) {
      aK0[h]     = p[r0 + h];       aK0[8 + h] = p[r1 + h];
      aK1[h]     = p[32 + r0 + h];  aK1[8 + h] = p[32 + r1 + h];
    }
  }
  // W2 B-fragments (K = hidden d, N = coord c = laneLo; cols >= 6 are zero).
  v16h bW0, bW1;
  {
    const int c = laneLo;
#pragma unroll
    for (int h = 0; h < 16; ++h) {
      const int K = (h < 8) ? (8 * laneHi + h) : (16 + 8 * laneHi + (h - 8));
      bW0[h] = (c < NC) ? (_Float16)aW2[K * NC + c]        : (_Float16)0.0f;
      bW1[h] = (c < NC) ? (_Float16)aW2[(K + 32) * NC + c] : (_Float16)0.0f;
    }
  }
  const float b2c = (laneLo < NC) ? aB2[laneLo] : 0.0f;

  v8f acc[NC][2];
  for (int c = 0; c < NC; ++c)
    for (int dt = 0; dt < 2; ++dt)
#pragma unroll
      for (int r = 0; r < 8; ++r) acc[c][dt][r] = 0.0f;

  for (int cc = 0; cc < 4; ++cc) {
    const int j0 = (cc * 8 + wave) * 32;

    // ---- scores + sigmoid for 32 j's ----
    for (int jj = 0; jj < 32; ++jj) {
      const int j = j0 + jj;
      const _Float16* p = qp + (size_t)(b * Nsz + j) * 64;
      __builtin_prefetch(qp + (size_t)(b * Nsz + j + 1) * 64, 0, 3);

      v16h a0, a1;   // mish(K' + Q') in A-matrix layout, f16
#pragma unroll
      for (int h = 0; h < 8; ++h) {
        a0[h]     = (_Float16)mishf((float)aK0[h]     + (float)p[r0 + h]);
        a0[8 + h] = (_Float16)mishf((float)aK0[8 + h] + (float)p[r1 + h]);
        a1[h]     = (_Float16)mishf((float)aK1[h]     + (float)p[32 + r0 + h]);
        a1[8 + h] = (_Float16)mishf((float)aK1[8 + h] + (float)p[32 + r1 + h]);
      }
      v8f s;
#pragma unroll
      for (int r = 0; r < 8; ++r) s[r] = 0.0f;
      s = __builtin_amdgcn_wmma_f32_16x16x32_f16(false, a0, false, bW0,
                                                 (short)0, s, false, false);
      s = __builtin_amdgcn_wmma_f32_16x16x32_f16(false, a1, false, bW1,
                                                 (short)0, s, false, false);
      if (laneLo < NC) {
        const int c = laneLo;
#pragma unroll
        for (int r = 0; r < 8; ++r) {
          const int iL = r + 8 * laneHi;
          float sv = s[r] + b2c;
          if (i0 + iL == j) sv -= 10000.0f;    // eye(n)*MASK_CONST
          wlds[wave][c][iL][jj] = (_Float16)sigmoidf_(sv);
        }
      }
    }
    asm volatile("s_wait_dscnt 0" ::: "memory");  // per-wave LDS RAW fence

    // ---- V B-fragments for this chunk (K = jj, N = d) ----
    v16h bV0, bV1;
    {
      const _Float16* pv0 = vh + ((size_t)b * 32 + laneLo)      * Nsz + j0 + 8 * laneHi;
      const _Float16* pv1 = vh + ((size_t)b * 32 + 16 + laneLo) * Nsz + j0 + 8 * laneHi;
#pragma unroll
      for (int h = 0; h < 16; ++h) { const int o = (h < 8) ? h : h + 8; bV0[h] = pv0[o]; bV1[h] = pv1[o]; }
    }

    // ---- agg accumulation: 6 coords x 2 d-tiles ----
#pragma unroll
    for (int c = 0; c < NC; ++c) {
      v16h aw;   // w[i, jj] in A-matrix layout
#pragma unroll
      for (int h = 0; h < 8; ++h) {
        aw[h]     = wlds[wave][c][laneLo][r0 + h];
        aw[8 + h] = wlds[wave][c][laneLo][r1 + h];
      }
      acc[c][0] = __builtin_amdgcn_wmma_f32_16x16x32_f16(false, aw, false, bV0,
                                                         (short)0, acc[c][0], false, false);
      acc[c][1] = __builtin_amdgcn_wmma_f32_16x16x32_f16(false, aw, false, bV1,
                                                         (short)0, acc[c][1], false, false);
    }
  }

  // ---- cross-wave reduction (ds_add_f32) ----
  for (int c = 0; c < NC; ++c)
    for (int dt = 0; dt < 2; ++dt)
#pragma unroll
      for (int r = 0; r < 8; ++r) {
        const int iL = r + 8 * laneHi;
        const int d  = dt * 16 + laneLo;
        atomicAdd(&aggL[c][iL][d], acc[c][dt][r]);
      }
  __syncthreads();

  for (int idx = tid; idx < NC * 16 * 32; idx += 256) {
    const int c  = idx >> 9;
    const int iL = (idx >> 5) & 15;
    const int d  = idx & 31;
    agg_g[((size_t)c * BN + (b * Nsz + i0 + iL)) * 32 + d] = aggL[c][iL][d];
  }
}

// ---------------------------------------------------------------------------
// Kernel 3a: collapse the purely-linear decoder:
//   wmu[c][k]  = sum_o decW[c][k,o] * muW[c][o]   (k < 64)
//   cmu[c]     = muB[c] + decB[c] . muW[c]        (same for sig)
// ---------------------------------------------------------------------------
__global__ __launch_bounds__(256) void collapse_kernel(
    DecParams P, float* __restrict__ wmu, float* __restrict__ cmu,
    float* __restrict__ wsig, float* __restrict__ csig) {
  const int t = blockIdx.x * 256 + threadIdx.x;
  if (t >= NC * 64) return;
  const int c = t >> 6, k = t & 63;
  const float* dW = P.decW[c];
  const float* mW = P.muW[c];
  const float* sW = P.sigW[c];
  float sm = 0.0f, ss = 0.0f;
#pragma unroll 8
  for (int o = 0; o < 128; ++o) {
    const float w = dW[k * 128 + o];
    sm += w * mW[o];
    ss += w * sW[o];
  }
  wmu[c * 64 + k] = sm;
  wsig[c * 64 + k] = ss;
  if (k == 0) {
    const float* dB = P.decB[c];
    float cm = P.muB[c][0], cs = P.sigB[c][0];
#pragma unroll 8
    for (int o = 0; o < 128; ++o) { cm += dB[o] * mW[o]; cs += dB[o] * sW[o]; }
    cmu[c] = cm;
    csig[c] = cs;
  }
}

// ---------------------------------------------------------------------------
// Kernel 3b: per (coord, token): mu = cmu + [v|agg].wmu ; sig = sigmoid(...)+EPS
// Output layout: (mu_x, sig_x, mu_y, sig_y, ...), each [B,N] flat.
// ---------------------------------------------------------------------------
__global__ __launch_bounds__(256) void dec_kernel(
    const float* __restrict__ vbuf, const float* __restrict__ agg_g,
    const float* __restrict__ wmu, const float* __restrict__ cmu,
    const float* __restrict__ wsig, const float* __restrict__ csig,
    float* __restrict__ out) {
  const int gid = blockIdx.x * 256 + threadIdx.x;
  if (gid >= NC * BN) return;
  const int c = gid / BN, t = gid % BN;
  float am = cmu[c], as = csig[c];
  const float* wm = wmu + c * 64;
  const float* ws = wsig + c * 64;
  const float* vv = vbuf + (size_t)t * 32;
  const float* ag = agg_g + ((size_t)c * BN + t) * 32;
#pragma unroll 8
  for (int k = 0; k < 32; ++k) {
    const float v = vv[k];
    am += v * wm[k];
    as += v * ws[k];
  }
#pragma unroll 8
  for (int k = 0; k < 32; ++k) {
    const float a = ag[k];
    am += a * wm[32 + k];
    as += a * ws[32 + k];
  }
  out[(size_t)(2 * c) * BN + t]     = am;
  out[(size_t)(2 * c + 1) * BN + t] = sigmoidf_(as) + 1e-7f;  // EPS = 1e-6/10
}

// ---------------------------------------------------------------------------
// Launcher. Input order (setup_inputs insertion order):
//  0: x
//  1..4:  key  W1[10,64] b1[64] W2[64,32] b2[32]
//  5..8:  query (same)
//  9..12: value (same)
// 13..16: att  W1[64,64] b1[64] W2[64,6] b2[6]
// 17+6*ci .. : per coord (x,y,z,vx,vy,vz): decW[64,128] decB[128]
//              muW[128,1] muB[1] sigW[128,1] sigB[1]
// ---------------------------------------------------------------------------
extern "C" void kernel_launch(void* const* d_in, const int* in_sizes, int n_in,
                              void* d_out, int out_size, void* d_ws, size_t ws_size,
                              hipStream_t stream) {
  (void)in_sizes; (void)n_in; (void)out_size; (void)ws_size;
  char* ws = (char*)d_ws;

  // Workspace layout (bytes):
  _Float16* kp   = (_Float16*)(ws + 0);          //  262144: K' f16 [B*N][64]
  _Float16* qp   = (_Float16*)(ws + 262144);     //  262144: Q' f16 [B*N][64]
  _Float16* vh   = (_Float16*)(ws + 524288);     //  131072: v^T f16 [B][32][N]
  float*    vbuf = (float*)   (ws + 655360);     //  262144: v f32 [B*N][32]
  float*    agg  = (float*)   (ws + 917504);     // 1572864: agg f32 [6][B*N][32]
  float*    wmu  = (float*)   (ws + 2490368);    //    1536
  float*    cmu  = (float*)   (ws + 2491904);    //      32
  float*    wsig = (float*)   (ws + 2491936);    //    1536
  float*    csig = (float*)   (ws + 2493472);    //      32

  EncParams ep;
  ep.x   = (const float*)d_in[0];
  ep.kW1 = (const float*)d_in[1];  ep.kB1 = (const float*)d_in[2];
  ep.kW2 = (const float*)d_in[3];  ep.kB2 = (const float*)d_in[4];
  ep.qW1 = (const float*)d_in[5];  ep.qB1 = (const float*)d_in[6];
  ep.qW2 = (const float*)d_in[7];  ep.qB2 = (const float*)d_in[8];
  ep.vW1 = (const float*)d_in[9];  ep.vB1 = (const float*)d_in[10];
  ep.vW2 = (const float*)d_in[11]; ep.vB2 = (const float*)d_in[12];
  ep.aW1 = (const float*)d_in[13]; ep.aB1 = (const float*)d_in[14];

  DecParams dp;
  for (int ci = 0; ci < NC; ++ci) {
    const int base = 17 + ci * 6;
    dp.decW[ci] = (const float*)d_in[base + 0];
    dp.decB[ci] = (const float*)d_in[base + 1];
    dp.muW[ci]  = (const float*)d_in[base + 2];
    dp.muB[ci]  = (const float*)d_in[base + 3];
    dp.sigW[ci] = (const float*)d_in[base + 4];
    dp.sigB[ci] = (const float*)d_in[base + 5];
  }

  enc_kernel<<<(BN + 255) / 256, 256, 0, stream>>>(ep, kp, qp, vh, vbuf);
  attn_kernel<<<Bsz * (Nsz / 16), 256, 0, stream>>>(
      kp, qp, vh, (const float*)d_in[15], (const float*)d_in[16], agg);
  collapse_kernel<<<2, 256, 0, stream>>>(dp, wmu, cmu, wsig, csig);
  dec_kernel<<<(NC * BN + 255) / 256, 256, 0, stream>>>(
      vbuf, agg, wmu, cmu, wsig, csig, (float*)d_out);
}
